// CrystalHypergraphConv_74071005987562
// MI455X (gfx1250) — compile-verified, matrix operand order checked by reading, not run on hardware
//
#include <hip/hip_runtime.h>
#include <hip/hip_bf16.h>

#define N_NODES  100000
#define N_HEDGES 200000
#define N_EDGES  1200000
#define N_GRAPHS 512
#define HDIM     64
#define HEDIM    35
#define HOUT     128
#define NLAYERS  3
#define KPAD     99           // real K for conv GEMMs
#define XDIM     92
#define XPAD     96           // x padded to 3 k-steps
#define MSGPAD   128          // msg padded to 4 k-steps
#define EPS      1e-5f

typedef __attribute__((ext_vector_type(16))) _Float16 v16h;
typedef __attribute__((ext_vector_type(8)))  _Float16 v8h;
typedef __attribute__((ext_vector_type(8)))  float    v8f;

__device__ __forceinline__ float dsig(float x) { return 1.0f / (1.0f + __expf(-x)); }
__device__ __forceinline__ float dsp(float x)  { return fmaxf(x, 0.0f) + log1pf(__expf(-fabsf(x))); }

// ---------------------------------------------------------------- utilities
__global__ void k_fill(float* __restrict__ p, long n) {
    long i = (long)blockIdx.x * blockDim.x + threadIdx.x;
    if (i < n) p[i] = 0.0f;
}

__global__ void k_counts(const int* __restrict__ nidx, const int* __restrict__ hidx,
                         float* __restrict__ ncnt, float* __restrict__ hcnt, int e) {
    int i = blockIdx.x * blockDim.x + threadIdx.x;
    if (i < e) {
        atomicAdd(&ncnt[nidx[i]], 1.0f);
        atomicAdd(&hcnt[hidx[i]], 1.0f);
    }
}

// x[N,92] f32 -> x16[N,96] f16 zero-padded
__global__ void k_x16(const float* __restrict__ x, unsigned short* __restrict__ x16u) {
    long idx = (long)blockIdx.x * blockDim.x + threadIdx.x;
    if (idx >= (long)N_NODES * XPAD) return;
    long row = idx / XPAD;
    int k = (int)(idx % XPAD);
    float v = (k < XDIM) ? x[row * XDIM + k] : 0.0f;
    ((_Float16*)x16u)[idx] = (_Float16)v;
}

// Pack W[K][N] (row-major f32) into WMMA B-fragment order, f16, zero-padded.
// out element idx = (((ks*NT+nt)*32)+lane)*16 + e
// k = ks*32 + (e<8?0:16) + 8*(lane>>4) + (e&7);  n = nt*16 + (lane&15)
__global__ void k_pack(const float* __restrict__ W, unsigned short* __restrict__ outp,
                       int kreal, int nreal, int nt_cnt, int ks_cnt) {
    int idx = blockIdx.x * blockDim.x + threadIdx.x;
    int total = ks_cnt * nt_cnt * 512;
    if (idx >= total) return;
    int e    = idx & 15;
    int lane = (idx >> 4) & 31;
    int blk  = idx >> 9;                  // ks*NT + nt
    int nt   = blk % nt_cnt;
    int ks   = blk / nt_cnt;
    int k = ks * 32 + ((e < 8) ? 0 : 16) + 8 * (lane >> 4) + (e & 7);
    int n = nt * 16 + (lane & 15);
    float v = (k < kreal && n < nreal) ? W[(long)k * nreal + n] : 0.0f;
    ((_Float16*)outp)[idx] = (_Float16)v;
}

// ---------------------------------------------------------------- embed: h = x16 @ We + be  (K=96 -> 3 ksteps)
__global__ void __launch_bounds__(32)
k_embed(const unsigned short* __restrict__ x16u, const unsigned short* __restrict__ pWe_u,
        const float* __restrict__ be, float* __restrict__ h, unsigned short* __restrict__ h16u) {
    const _Float16* x16 = (const _Float16*)x16u;
    const v16h* pB = (const v16h*)pWe_u;
    _Float16* h16 = (_Float16*)h16u;
    int tile = blockIdx.x;
    int lane = threadIdx.x;
    int rlo = lane & 15, hl = lane >> 4;
    long rowbase = ((long)tile * 16 + rlo) * XPAD;
    v8f acc[4] = {};
    for (int ks = 0; ks < 3; ++ks) {
        int k1 = ks * 32 + 8 * hl;
        v8h r1 = *(const v8h*)(x16 + rowbase + k1);
        v8h r2 = *(const v8h*)(x16 + rowbase + k1 + 16);
        v16h a;
#pragma unroll
        for (int i = 0; i < 8; ++i) { a[i] = r1[i]; a[8 + i] = r2[i]; }
#pragma unroll
        for (int nt = 0; nt < 4; ++nt) {
            v16h b = pB[(ks * 4 + nt) * 32 + lane];
            acc[nt] = __builtin_amdgcn_wmma_f32_16x16x32_f16(false, a, false, b,
                                                             (short)0, acc[nt], false, false);
        }
    }
#pragma unroll
    for (int nt = 0; nt < 4; ++nt)
#pragma unroll
        for (int q = 0; q < 8; ++q) {
            int m = q + 8 * hl;
            int col = nt * 16 + rlo;
            long o = ((long)tile * 16 + m) * HDIM + col;
            float v = acc[nt][q] + be[col];
            h[o] = v;
            h16[o] = (_Float16)v;
        }
}

// ---------------------------------------------------------------- scatter-sum h[node] into hedge accumulators
__global__ void k_edge_gather(const int* __restrict__ nidx, const int* __restrict__ hidx,
                              const float* __restrict__ h, float* __restrict__ agg) {
    long idx = (long)blockIdx.x * blockDim.x + threadIdx.x;
    if (idx >= (long)N_EDGES * HDIM) return;
    long e = idx >> 6;
    int c = (int)(idx & 63);
    atomicAdd(&agg[(long)hidx[e] * HDIM + c], h[(long)nidx[e] * HDIM + c]);
}

// ---------------------------------------------------------------- msg16[NH,128] = [agg/cnt (64) | hattr (35) | 0...]
__global__ void k_hmsg(const float* __restrict__ agg, const float* __restrict__ hcnt,
                       const float* __restrict__ hattr, unsigned short* __restrict__ msg16u) {
    long idx = (long)blockIdx.x * blockDim.x + threadIdx.x;
    if (idx >= (long)N_HEDGES * MSGPAD) return;
    long row = idx >> 7;
    int k = (int)(idx & 127);
    float v;
    if (k < HDIM)      v = agg[row * HDIM + k] / fmaxf(hcnt[row], 1.0f);
    else if (k < KPAD) v = hattr[row * HEDIM + (k - HDIM)];
    else               v = 0.0f;
    ((_Float16*)msg16u)[idx] = (_Float16)v;
}

// ---------------------------------------------------------------- hedge GEMM: msg16[NH,128] @ {Wf1,Wc1}
__global__ void __launch_bounds__(32)
k_hedge_gemm(const unsigned short* __restrict__ msg16u,
             const unsigned short* __restrict__ pWf_u, const unsigned short* __restrict__ pWc_u,
             const float* __restrict__ bf1, const float* __restrict__ bc1,
             float* __restrict__ zf, float* __restrict__ zc) {
    const _Float16* msg = (const _Float16*)msg16u;
    const v16h* pF = (const v16h*)pWf_u;
    const v16h* pC = (const v16h*)pWc_u;
    int tile = blockIdx.x;
    int lane = threadIdx.x;
    int rlo = lane & 15, hl = lane >> 4;
    long rowbase = ((long)tile * 16 + rlo) * MSGPAD;
    v8f accF[3] = {}, accC[3] = {};
    for (int ks = 0; ks < 4; ++ks) {
        int k1 = ks * 32 + 8 * hl;
        v8h r1 = *(const v8h*)(msg + rowbase + k1);
        v8h r2 = *(const v8h*)(msg + rowbase + k1 + 16);
        v16h a;
#pragma unroll
        for (int i = 0; i < 8; ++i) { a[i] = r1[i]; a[8 + i] = r2[i]; }
#pragma unroll
        for (int nt = 0; nt < 3; ++nt) {
            v16h bF = pF[(ks * 3 + nt) * 32 + lane];
            accF[nt] = __builtin_amdgcn_wmma_f32_16x16x32_f16(false, a, false, bF,
                                                              (short)0, accF[nt], false, false);
            v16h bC = pC[(ks * 3 + nt) * 32 + lane];
            accC[nt] = __builtin_amdgcn_wmma_f32_16x16x32_f16(false, a, false, bC,
                                                              (short)0, accC[nt], false, false);
        }
    }
#pragma unroll
    for (int nt = 0; nt < 3; ++nt)
#pragma unroll
        for (int q = 0; q < 8; ++q) {
            int m = q + 8 * hl;
            int col = nt * 16 + rlo;
            if (col < HEDIM) {
                long o = ((long)tile * 16 + m) * HEDIM + col;
                zf[o] = accF[nt][q] + bf1[col];
                zc[o] = accC[nt][q] + bc1[col];
            }
        }
}

// ---------------------------------------------------------------- BN stats over hedges (35 cols, 2 tensors)
__global__ void __launch_bounds__(64)
k_hstats(const float* __restrict__ zf, const float* __restrict__ zc,
         float* __restrict__ stats, int rows_per_blk) {
    int c = threadIdx.x;
    if (c >= HEDIM) return;
    long r0 = (long)blockIdx.x * rows_per_blk;
    long r1 = r0 + rows_per_blk; if (r1 > N_HEDGES) r1 = N_HEDGES;
    float sF = 0, qF = 0, sC = 0, qC = 0;
    for (long r = r0; r < r1; ++r) {
        float a = zf[r * HEDIM + c]; sF += a; qF += a * a;
        float b = zc[r * HEDIM + c]; sC += b; qC += b * b;
    }
    atomicAdd(&stats[c], sF);
    atomicAdd(&stats[HEDIM + c], qF);
    atomicAdd(&stats[2 * HEDIM + c], sC);
    atomicAdd(&stats[3 * HEDIM + c], qC);
}

// ---------------------------------------------------------------- hedge_feats = sigmoid(bn(zf))*softplus(bn(zc)), padded to 64 f16
__global__ void k_hfeats(const float* __restrict__ zf, const float* __restrict__ zc,
                         const float* __restrict__ stats,
                         const float* __restrict__ gf, const float* __restrict__ btf,
                         const float* __restrict__ gc, const float* __restrict__ btc,
                         unsigned short* __restrict__ hf16u) {
    long idx = (long)blockIdx.x * blockDim.x + threadIdx.x;
    if (idx >= (long)N_HEDGES * HDIM) return;
    long row = idx >> 6;
    int c = (int)(idx & 63);
    float v = 0.0f;
    if (c < HEDIM) {
        const float inv = 1.0f / (float)N_HEDGES;
        float mF = stats[c] * inv;
        float vF = stats[HEDIM + c] * inv - mF * mF;
        float mC = stats[2 * HEDIM + c] * inv;
        float vC = stats[3 * HEDIM + c] * inv - mC * mC;
        float xf = gf[c] * (zf[row * HEDIM + c] - mF) * rsqrtf(vF + EPS) + btf[c];
        float xc = gc[c] * (zc[row * HEDIM + c] - mC) * rsqrtf(vC + EPS) + btc[c];
        v = dsig(xf) * dsp(xc);
    }
    ((_Float16*)hf16u)[idx] = (_Float16)v;
}

// ---------------------------------------------------------------- edge GEMM + fused activation + scatter-add
// z[e] = [h16[node] (K 0..63) | hf16[hedge] (K 64..127, zero past 98)]
__global__ void __launch_bounds__(32)
k_edge_gemm(const unsigned short* __restrict__ h16u, const unsigned short* __restrict__ hf16u,
            const int* __restrict__ nidx, const int* __restrict__ hidx,
            const unsigned short* __restrict__ pWf_u, const unsigned short* __restrict__ pWc_u,
            const float* __restrict__ bf2, const float* __restrict__ bc2,
            float* __restrict__ hnext) {
    const _Float16* h16  = (const _Float16*)h16u;
    const _Float16* hf16 = (const _Float16*)hf16u;
    const v16h* pF = (const v16h*)pWf_u;
    const v16h* pC = (const v16h*)pWc_u;
    int tile = blockIdx.x;
    int lane = threadIdx.x;
    int rlo = lane & 15, hl = lane >> 4;
    long e = (long)tile * 16 + rlo;
    long nd = nidx[e], hg = hidx[e];
    v8f accF[4] = {}, accC[4] = {};
    for (int ks = 0; ks < 4; ++ks) {
        int k1 = ks * 32 + 8 * hl;
        int k2 = k1 + 16;
        v8h r1 = (k1 < HDIM) ? *(const v8h*)(h16 + nd * HDIM + k1)
                             : *(const v8h*)(hf16 + hg * HDIM + (k1 - HDIM));
        v8h r2 = (k2 < HDIM) ? *(const v8h*)(h16 + nd * HDIM + k2)
                             : *(const v8h*)(hf16 + hg * HDIM + (k2 - HDIM));
        v16h a;
#pragma unroll
        for (int i = 0; i < 8; ++i) { a[i] = r1[i]; a[8 + i] = r2[i]; }
#pragma unroll
        for (int nt = 0; nt < 4; ++nt) {
            v16h bF = pF[(ks * 4 + nt) * 32 + lane];
            accF[nt] = __builtin_amdgcn_wmma_f32_16x16x32_f16(false, a, false, bF,
                                                              (short)0, accF[nt], false, false);
            v16h bC = pC[(ks * 4 + nt) * 32 + lane];
            accC[nt] = __builtin_amdgcn_wmma_f32_16x16x32_f16(false, a, false, bC,
                                                              (short)0, accC[nt], false, false);
        }
    }
    long ndr[8];
#pragma unroll
    for (int q = 0; q < 8; ++q) ndr[q] = nidx[(long)tile * 16 + q + 8 * hl];
#pragma unroll
    for (int nt = 0; nt < 4; ++nt) {
        int col = nt * 16 + rlo;
        float bF = bf2[col], bC = bc2[col];
#pragma unroll
        for (int q = 0; q < 8; ++q) {
            float r = dsig(accF[nt][q] + bF) * dsp(accC[nt][q] + bC);
            atomicAdd(&hnext[ndr[q] * HDIM + col], r);
        }
    }
}

// ---------------------------------------------------------------- BN stats over nodes (64 cols, on hnext/cnt)
__global__ void __launch_bounds__(64)
k_nstats(const float* __restrict__ hnext, const float* __restrict__ ncnt,
         float* __restrict__ stats, int rows_per_blk) {
    int c = threadIdx.x;
    long r0 = (long)blockIdx.x * rows_per_blk;
    long r1 = r0 + rows_per_blk; if (r1 > N_NODES) r1 = N_NODES;
    float s = 0, q = 0;
    for (long r = r0; r < r1; ++r) {
        float v = hnext[r * HDIM + c] / fmaxf(ncnt[r], 1.0f);
        s += v; q += v * v;
    }
    atomicAdd(&stats[c], s);
    atomicAdd(&stats[HDIM + c], q);
}

// ---------------------------------------------------------------- h = relu(softplus(bn(mean) + h))
__global__ void k_nfinal(float* __restrict__ h, unsigned short* __restrict__ h16u,
                         const float* __restrict__ hnext, const float* __restrict__ ncnt,
                         const float* __restrict__ stats,
                         const float* __restrict__ go, const float* __restrict__ bto) {
    long idx = (long)blockIdx.x * blockDim.x + threadIdx.x;
    if (idx >= (long)N_NODES * HDIM) return;
    long r = idx >> 6;
    int c = (int)(idx & 63);
    const float inv = 1.0f / (float)N_NODES;
    float m = stats[c] * inv;
    float var = stats[HDIM + c] * inv - m * m;
    float v = hnext[idx] / fmaxf(ncnt[r], 1.0f);
    float bn = go[c] * (v - m) * rsqrtf(var + EPS) + bto[c];
    float y = fmaxf(dsp(bn + h[idx]), 0.0f);
    h[idx] = y;
    ((_Float16*)h16u)[idx] = (_Float16)y;
}

// ---------------------------------------------------------------- graph pooling + head
__global__ void k_pool(const int* __restrict__ batch, const float* __restrict__ h,
                       float* __restrict__ gsum) {
    long idx = (long)blockIdx.x * blockDim.x + threadIdx.x;
    if (idx >= (long)N_NODES * HDIM) return;
    long r = idx >> 6;
    int c = (int)(idx & 63);
    atomicAdd(&gsum[(long)batch[r] * HDIM + c], h[idx]);
}

__global__ void k_poolcnt(const int* __restrict__ batch, float* __restrict__ gcnt) {
    int i = blockIdx.x * blockDim.x + threadIdx.x;
    if (i < N_NODES) atomicAdd(&gcnt[batch[i]], 1.0f);
}

__global__ void __launch_bounds__(128)
k_g2(const float* __restrict__ gsum, const float* __restrict__ gcnt,
     const float* __restrict__ Wl2, const float* __restrict__ bl2, float* __restrict__ g2) {
    int gi = blockIdx.x, j = threadIdx.x;
    float invc = 1.0f / fmaxf(gcnt[gi], 1.0f);
    float acc = bl2[j];
    for (int k = 0; k < HDIM; ++k)
        acc += gsum[gi * HDIM + k] * invc * Wl2[k * HOUT + j];
    g2[gi * HOUT + j] = dsp(acc);
}

__global__ void k_out(const float* __restrict__ g2, const float* __restrict__ Wo,
                      const float* __restrict__ bo, float* __restrict__ outp) {
    int gi = blockIdx.x * blockDim.x + threadIdx.x;
    if (gi >= N_GRAPHS) return;
    float acc = bo[0];
    for (int k = 0; k < HOUT; ++k) acc += g2[gi * HOUT + k] * Wo[k];
    outp[gi] = acc;
}

// ================================================================ host
extern "C" void kernel_launch(void* const* d_in, const int* in_sizes, int n_in,
                              void* d_out, int out_size, void* d_ws, size_t ws_size,
                              hipStream_t stream) {
    const float* x      = (const float*)d_in[0];
    const int*   nidx   = (const int*)d_in[1];
    const int*   hidx   = nidx + N_EDGES;
    const float* hattr  = (const float*)d_in[2];
    const int*   batch  = (const int*)d_in[3];
    const float* We     = (const float*)d_in[4];
    const float* be     = (const float*)d_in[5];
    const float* Wf1    = (const float*)d_in[6];
    const float* bf1    = (const float*)d_in[7];
    const float* Wc1    = (const float*)d_in[8];
    const float* bc1    = (const float*)d_in[9];
    const float* Wf2    = (const float*)d_in[10];
    const float* bf2    = (const float*)d_in[11];
    const float* Wc2    = (const float*)d_in[12];
    const float* bc2    = (const float*)d_in[13];
    const float* bn_f_g = (const float*)d_in[14];
    const float* bn_f_b = (const float*)d_in[15];
    const float* bn_c_g = (const float*)d_in[16];
    const float* bn_c_b = (const float*)d_in[17];
    const float* bn_o_g = (const float*)d_in[18];
    const float* bn_o_b = (const float*)d_in[19];
    const float* Wl2    = (const float*)d_in[20];
    const float* bl2    = (const float*)d_in[21];
    const float* Wo     = (const float*)d_in[22];
    const float* bo     = (const float*)d_in[23];

    char* ws = (char*)d_ws;
    size_t off = 0;
    auto take = [&](size_t bytes) -> void* {
        void* p = ws + off;
        off = (off + bytes + 255) & ~(size_t)255;
        return p;
    };
    float* h        = (float*)take((size_t)N_NODES * HDIM * 4);
    float* hnext    = (float*)take((size_t)N_NODES * HDIM * 4);
    unsigned short* h16 = (unsigned short*)take((size_t)N_NODES * HDIM * 2);
    unsigned short* x16 = (unsigned short*)take((size_t)N_NODES * XPAD * 2);
    float* ncnt     = (float*)take((size_t)N_NODES * 4);
    float* agg      = (float*)take((size_t)N_HEDGES * HDIM * 4);
    unsigned short* msg16 = (unsigned short*)take((size_t)N_HEDGES * MSGPAD * 2);
    float* hcnt     = (float*)take((size_t)N_HEDGES * 4);
    float* zf       = (float*)take((size_t)N_HEDGES * HEDIM * 4);
    float* zc       = (float*)take((size_t)N_HEDGES * HEDIM * 4);
    unsigned short* hf16 = (unsigned short*)take((size_t)N_HEDGES * HDIM * 2);
    float* statsH   = (float*)take(4 * HEDIM * 4);
    float* statsO   = (float*)take(2 * HDIM * 4);
    float* gsum     = (float*)take((size_t)N_GRAPHS * HDIM * 4);
    float* gcnt     = (float*)take((size_t)N_GRAPHS * 4);
    float* g2       = (float*)take((size_t)N_GRAPHS * HOUT * 4);
    const int PWE  = 3 * 4 * 512;   // ksteps=3, ntiles=4
    const int PW1  = 4 * 3 * 512;   // ksteps=4, ntiles=3
    const int PW2  = 4 * 4 * 512;   // ksteps=4, ntiles=4
    unsigned short* pWe  = (unsigned short*)take((size_t)PWE * 2);
    unsigned short* pW1f = (unsigned short*)take((size_t)NLAYERS * PW1 * 2);
    unsigned short* pW1c = (unsigned short*)take((size_t)NLAYERS * PW1 * 2);
    unsigned short* pW2f = (unsigned short*)take((size_t)NLAYERS * PW2 * 2);
    unsigned short* pW2c = (unsigned short*)take((size_t)NLAYERS * PW2 * 2);

    auto blks = [](long n, int b) { return (unsigned)((n + b - 1) / b); };

    // degree counts + global-accumulator zeroing (once per launch)
    k_fill<<<blks(N_NODES, 256), 256, 0, stream>>>(ncnt, N_NODES);
    k_fill<<<blks(N_HEDGES, 256), 256, 0, stream>>>(hcnt, N_HEDGES);
    k_fill<<<blks((long)N_GRAPHS * HDIM, 256), 256, 0, stream>>>(gsum, (long)N_GRAPHS * HDIM);
    k_fill<<<blks(N_GRAPHS, 256), 256, 0, stream>>>(gcnt, N_GRAPHS);
    k_counts<<<blks(N_EDGES, 256), 256, 0, stream>>>(nidx, hidx, ncnt, hcnt, N_EDGES);

    // f16 staging of x + pack all weights into WMMA B-fragment order
    k_x16<<<blks((long)N_NODES * XPAD, 256), 256, 0, stream>>>(x, x16);
    k_pack<<<blks(PWE, 256), 256, 0, stream>>>(We, pWe, XDIM, HDIM, 4, 3);
    for (int l = 0; l < NLAYERS; ++l) {
        k_pack<<<blks(PW1, 256), 256, 0, stream>>>(Wf1 + (long)l * KPAD * HEDIM, pW1f + (long)l * PW1, KPAD, HEDIM, 3, 4);
        k_pack<<<blks(PW1, 256), 256, 0, stream>>>(Wc1 + (long)l * KPAD * HEDIM, pW1c + (long)l * PW1, KPAD, HEDIM, 3, 4);
        k_pack<<<blks(PW2, 256), 256, 0, stream>>>(Wf2 + (long)l * KPAD * HDIM, pW2f + (long)l * PW2, KPAD, HDIM, 4, 4);
        k_pack<<<blks(PW2, 256), 256, 0, stream>>>(Wc2 + (long)l * KPAD * HDIM, pW2c + (long)l * PW2, KPAD, HDIM, 4, 4);
    }

    // embed: h = x @ We + be
    k_embed<<<N_NODES / 16, 32, 0, stream>>>(x16, pWe, be, h, h16);

    for (int l = 0; l < NLAYERS; ++l) {
        k_fill<<<blks((long)N_HEDGES * HDIM, 256), 256, 0, stream>>>(agg, (long)N_HEDGES * HDIM);
        k_fill<<<blks((long)N_NODES * HDIM, 256), 256, 0, stream>>>(hnext, (long)N_NODES * HDIM);
        k_fill<<<1, 256, 0, stream>>>(statsH, 4 * HEDIM);
        k_fill<<<1, 256, 0, stream>>>(statsO, 2 * HDIM);

        k_edge_gather<<<blks((long)N_EDGES * HDIM, 256), 256, 0, stream>>>(nidx, hidx, h, agg);

        k_hmsg<<<blks((long)N_HEDGES * MSGPAD, 256), 256, 0, stream>>>(agg, hcnt, hattr, msg16);

        k_hedge_gemm<<<N_HEDGES / 16, 32, 0, stream>>>(
            msg16, pW1f + (long)l * PW1, pW1c + (long)l * PW1,
            bf1 + l * HEDIM, bc1 + l * HEDIM, zf, zc);

        k_hstats<<<blks(N_HEDGES, 250), 64, 0, stream>>>(zf, zc, statsH, 250);

        k_hfeats<<<blks((long)N_HEDGES * HDIM, 256), 256, 0, stream>>>(
            zf, zc, statsH, bn_f_g + l * HEDIM, bn_f_b + l * HEDIM,
            bn_c_g + l * HEDIM, bn_c_b + l * HEDIM, hf16);

        k_edge_gemm<<<N_EDGES / 16, 32, 0, stream>>>(
            h16, hf16, nidx, hidx, pW2f + (long)l * PW2, pW2c + (long)l * PW2,
            bf2 + l * HDIM, bc2 + l * HDIM, hnext);

        k_nstats<<<blks(N_NODES, 250), 64, 0, stream>>>(hnext, ncnt, statsO, 250);

        k_nfinal<<<blks((long)N_NODES * HDIM, 256), 256, 0, stream>>>(
            h, h16, hnext, ncnt, statsO, bn_o_g + l * HDIM, bn_o_b + l * HDIM);
    }

    // pooling + head
    k_pool<<<blks((long)N_NODES * HDIM, 256), 256, 0, stream>>>(batch, h, gsum);
    k_poolcnt<<<blks(N_NODES, 256), 256, 0, stream>>>(batch, gcnt);
    k_g2<<<N_GRAPHS, 128, 0, stream>>>(gsum, gcnt, Wl2, bl2, g2);
    k_out<<<blks(N_GRAPHS, 256), 256, 0, stream>>>(g2, Wo, bo, (float*)d_out);
}
